// OFTInjectedLinear_75007308857679
// MI455X (gfx1250) — matrix operand order repaired and checked by reading, not compiled, not run on hardware
//
#include <hip/hip_runtime.h>

// MI455X / gfx1250, wave32. out = x @ orth @ W^T via 4 Householder row
// updates + one bf16 WMMA GEMM (f32 accumulate), with CDNA5 async
// global->LDS staging (ASYNCcnt) and LDS double buffering.
// Workspace layout (needs 48 MB):
//   Yb : 4096*2048 bf16 (x after Householder reflections)   = 16 MB
//   Wb : 8192*2048 bf16 (W converted)                       = 32 MB

typedef __attribute__((ext_vector_type(16))) __bf16        v16bf;
typedef __attribute__((ext_vector_type(8)))  float         v8f;
typedef __attribute__((ext_vector_type(8)))  unsigned short ushort8;

#define NDIM 2048   // K
#define MDIM 8192   // out cols
#define ROWS 4096   // B*S
#define RREF 4

static __device__ __forceinline__ unsigned short f32_to_bf16(float f) {
  union { float f; unsigned u; } v; v.f = f;
  unsigned rounding = 0x7FFFu + ((v.u >> 16) & 1u);   // round-to-nearest-even
  return (unsigned short)((v.u + rounding) >> 16);
}

// CDNA5 async global->LDS copy, 16B per lane, tracked by ASYNCcnt.
// VDST operand = LDS byte address (low 32 bits of the generic pointer per the
// flat/LDS aperture rule), VADDR = 64-bit global address.
static __device__ __forceinline__ void async_copy_b128(
    unsigned short* lds_dst, const unsigned short* g_src) {
  unsigned laddr = (unsigned)(unsigned long long)lds_dst;
  unsigned long long gaddr = (unsigned long long)g_src;
  asm volatile("global_load_async_to_lds_b128 %0, %1, off"
               :: "v"(laddr), "v"(gaddr)
               : "memory");
}

static __device__ __forceinline__ void wait_async0() {
  asm volatile("s_wait_asynccnt 0x0" ::: "memory");
}

// ---------------------------------------------------------------------------
// Kernel 1: one block per row of x. Apply r=4 Householder reflections
// sequentially (v -= 2*(v.R_j)*R_j), emit bf16 row into Yb.
// ---------------------------------------------------------------------------
__global__ __launch_bounds__(256) void householder_rows_kernel(
    const float* __restrict__ x, const float* __restrict__ R,
    unsigned short* __restrict__ Yb) {
  const int row  = blockIdx.x;
  const int t    = threadIdx.x;
  const int lane = t & 31;
  const int wave = t >> 5;

  float v[8];
  float rv[RREF][8];
  const float* xrow = x + (size_t)row * NDIM;
#pragma unroll
  for (int i = 0; i < 8; ++i) v[i] = xrow[i * 256 + t];
#pragma unroll
  for (int j = 0; j < RREF; ++j)
#pragma unroll
    for (int i = 0; i < 8; ++i) rv[j][i] = R[j * NDIM + i * 256 + t];

  __shared__ float red[8];
  __shared__ float dotS;

#pragma unroll
  for (int j = 0; j < RREF; ++j) {
    float p = 0.f;
#pragma unroll
    for (int i = 0; i < 8; ++i) p += v[i] * rv[j][i];
    // wave32 reduction
#pragma unroll
    for (int off = 16; off > 0; off >>= 1) p += __shfl_xor(p, off, 32);
    if (lane == 0) red[wave] = p;
    __syncthreads();
    if (t == 0) {
      float s = 0.f;
#pragma unroll
      for (int w = 0; w < 8; ++w) s += red[w];
      dotS = s;
    }
    __syncthreads();
    const float c = 2.0f * dotS;
#pragma unroll
    for (int i = 0; i < 8; ++i) v[i] -= c * rv[j][i];
    __syncthreads();   // red[] reused next iteration
  }

#pragma unroll
  for (int i = 0; i < 8; ++i)
    Yb[(size_t)row * NDIM + i * 256 + t] = f32_to_bf16(v[i]);
}

// ---------------------------------------------------------------------------
// Kernel 2: convert W (f32) -> Wb (bf16), 8 elems/thread, vectorized store.
// ---------------------------------------------------------------------------
__global__ __launch_bounds__(256) void w_to_bf16_kernel(
    const float* __restrict__ W, unsigned short* __restrict__ Wb, int total) {
  int idx = (blockIdx.x * 256 + threadIdx.x) * 8;
  if (idx >= total) return;
  ushort8 o;
#pragma unroll
  for (int i = 0; i < 8; ++i) o[i] = f32_to_bf16(W[idx + i]);
  *(ushort8*)(Wb + idx) = o;
}

// ---------------------------------------------------------------------------
// Kernel 3: C[4096x8192] = Yb[4096x2048] * Wb^T, bf16 WMMA, f32 accumulate.
// Block tile 128x128, BK=64, 8 waves (2 along M x 4 along N), wave tile 64x32.
// Async global->LDS staging, double-buffered.
// ---------------------------------------------------------------------------
#define BM 128
#define BN 128
#define BK 64
#define LDK 72   // padded LDS row stride (bf16 elems) to spread banks

union FragU {
  v16bf   bf;
  ushort8 u8[2];
};

__global__ __launch_bounds__(256) void oft_gemm_bf16_kernel(
    const unsigned short* __restrict__ Yb,
    const unsigned short* __restrict__ Wb,
    float* __restrict__ C) {
  __shared__ unsigned short lsA[2][BM * LDK];   // 2 x 18 KB
  __shared__ unsigned short lsB[2][BN * LDK];   // 2 x 18 KB

  const int t    = threadIdx.x;
  const int lane = t & 31;
  const int wave = t >> 5;
  const int wm   = wave & 1;    // wave row (0..1): 64 M-rows each
  const int wn   = wave >> 1;   // wave col (0..3): 32 N-cols each
  const int mBase = blockIdx.y * BM;
  const int nBase = blockIdx.x * BN;

  // staging role: each thread owns half a 64-col row (32 bf16 = 4 x 16B)
  const int srow  = t >> 1;     // 0..127
  const int shalf = t & 1;      // 0..1

  const unsigned short* gA0 = Yb + (size_t)(mBase + srow) * NDIM + shalf * 32;
  const unsigned short* gB0 = Wb + (size_t)(nBase + srow) * NDIM + shalf * 32;
  const int sOff = srow * LDK + shalf * 32;

  v8f acc[4][2];
#pragma unroll
  for (int mi = 0; mi < 4; ++mi)
#pragma unroll
    for (int ni = 0; ni < 2; ++ni) acc[mi][ni] = {};

  const int ln = lane & 15;
  const int hi = lane >> 4;     // half-wave: 0 (lanes 0-15) / 1 (lanes 16-31)

  const int NT = NDIM / BK;     // 32 K-tiles

  // ---- prologue: async-stage tile 0 into buffer 0 ----
#pragma unroll
  for (int i = 0; i < 4; ++i) {
    async_copy_b128(&lsA[0][sOff + i * 8], gA0 + i * 8);
    async_copy_b128(&lsB[0][sOff + i * 8], gB0 + i * 8);
  }

  for (int it = 0; it < NT; ++it) {
    const int cur = it & 1;
    // own async loads for tile `it` have landed; barrier makes all landed
    // (and guarantees everyone finished computing on the other buffer).
    wait_async0();
    __syncthreads();

    // ---- issue async loads for tile it+1 into the other buffer ----
    if (it + 1 < NT) {
      const int nk  = (it + 1) * BK;
      const int nxt = cur ^ 1;
#pragma unroll
      for (int i = 0; i < 4; ++i) {
        async_copy_b128(&lsA[nxt][sOff + i * 8], gA0 + nk + i * 8);
        async_copy_b128(&lsB[nxt][sOff + i * 8], gB0 + nk + i * 8);
      }
    }

    // ---- compute on tile `it` (two K=32 WMMA steps) ----
    const unsigned short* sA = lsA[cur];
    const unsigned short* sB = lsB[cur];
#pragma unroll
    for (int kk = 0; kk < BK; kk += 32) {
      FragU a[4], b[2];
      // A 16x32 bf16 layout: lanes 0-15 hold K {0..7,16..23}, lanes 16-31
      // hold K {8..15,24..31}, row M = lane%16.
#pragma unroll
      for (int mi = 0; mi < 4; ++mi) {
        const unsigned short* p = sA + (wm * 64 + mi * 16 + ln) * LDK + kk + hi * 8;
        a[mi].u8[0] = *(const ushort8*)(p);
        a[mi].u8[1] = *(const ushort8*)(p + 16);
      }
      // B 32x16 bf16 layout: lanes 0-15 hold K {0..7,16..23}, lanes 16-31 hold
      // K {8..15,24..31}, col N = lane%16. W row n supplies B column n.
#pragma unroll
      for (int ni = 0; ni < 2; ++ni) {
        const unsigned short* p = sB + (wn * 32 + ni * 16 + ln) * LDK + kk + hi * 8;
        b[ni].u8[0] = *(const ushort8*)(p);
        b[ni].u8[1] = *(const ushort8*)(p + 16);
      }
#pragma unroll
      for (int mi = 0; mi < 4; ++mi)
#pragma unroll
        for (int ni = 0; ni < 2; ++ni)
          acc[mi][ni] = __builtin_amdgcn_wmma_f32_16x16x32_bf16(
              false, a[mi].bf, false, b[ni].bf,
              (short)0, acc[mi][ni], false, false);
    }
  }

  // -------- write C per the 16x16 f32 C/D layout --------
  const int mAdd = hi * 8;
#pragma unroll
  for (int mi = 0; mi < 4; ++mi) {
#pragma unroll
    for (int ni = 0; ni < 2; ++ni) {
      float* cp = C + (size_t)(mBase + wm * 64 + mi * 16 + mAdd) * MDIM
                    + nBase + wn * 32 + ni * 16 + ln;
#pragma unroll
      for (int v = 0; v < 8; ++v) cp[(size_t)v * MDIM] = acc[mi][ni][v];
    }
  }
}

// ---------------------------------------------------------------------------
extern "C" void kernel_launch(void* const* d_in, const int* in_sizes, int n_in,
                              void* d_out, int out_size, void* d_ws, size_t ws_size,
                              hipStream_t stream) {
  const float* x = (const float*)d_in[0];   // [2,2048,2048]
  const float* R = (const float*)d_in[1];   // [4,2048,1]
  const float* W = (const float*)d_in[2];   // [8192,2048]
  float* out = (float*)d_out;               // [2,2048,8192]

  unsigned short* Yb = (unsigned short*)d_ws;                 // 16 MB
  unsigned short* Wb = Yb + (size_t)ROWS * NDIM;              // +32 MB

  householder_rows_kernel<<<ROWS, 256, 0, stream>>>(x, R, Yb);

  const int wTotal = MDIM * NDIM;
  w_to_bf16_kernel<<<wTotal / (256 * 8), 256, 0, stream>>>(W, Wb, wTotal);

  dim3 grid(MDIM / BN, ROWS / BM);   // 64 x 32
  oft_gemm_bf16_kernel<<<grid, 256, 0, stream>>>(Yb, Wb, out);
}
